// GConvLSTM_31756988186753
// MI455X (gfx1250) — compile-verified, matrix operand order checked by reading, not run on hardware
//
#include <hip/hip_runtime.h>
#include <math.h>
#include <stdint.h>

#define NN 100000
#define NE 1600000
#define LDA 40   // padded bf16 row stride in LDS: 16 DW data + 4 DW pad = 80 bytes

typedef __attribute__((ext_vector_type(16))) __bf16 v16bf;
typedef __attribute__((ext_vector_type(8)))  __bf16 v8bf;
typedef __attribute__((ext_vector_type(8)))  float  v8f;
typedef __attribute__((ext_vector_type(4)))  unsigned int u32x4;
typedef __attribute__((ext_vector_type(8)))  int i32x8;
typedef __attribute__((ext_vector_type(4)))  int i32x4;

__device__ __forceinline__ float sigmoidf(float x) { return 1.0f / (1.0f + __expf(-x)); }

// ---------------- Tensor Data Mover: 2D dword-tile load to LDS ----------------
// D# per CDNA5 ISA 8.3/8.4: group0 = {count/lds_addr/global_addr/type},
// group1 = {data_size=4B, pad_enable, pad_interval=3 (16 DW), pad_amount=3 (4 DW),
//           tensor_dim0/1, tile_dim0/1, tensor_dim0_stride}.
__device__ __forceinline__ void tdm_load_2d(const void* gptr, unsigned lds_off,
                                            unsigned tdim0_dw, unsigned tdim1,
                                            unsigned tile0_dw, unsigned tile1,
                                            unsigned stride_dw) {
    unsigned long long ga = (unsigned long long)gptr;
    u32x4 g0;
    g0[0] = 1u;                                            // count=1 (valid user D#)
    g0[1] = lds_off;                                       // lds_addr (bytes)
    g0[2] = (unsigned)(ga & 0xFFFFFFFFu);                  // global_addr[31:0]
    g0[3] = (unsigned)((ga >> 32) & 0x01FFFFFFu) | (2u << 30); // addr[56:32] | type=2
    i32x8 g1;
    g1[0] = (int)((2u << 16) | (1u << 20) | (3u << 22) | (3u << 25)); // 4B elems, pad 4DW/16DW
    g1[1] = (int)((tdim0_dw & 0xFFFFu) << 16);             // tensor_dim0[15:0]
    g1[2] = (int)(((tdim0_dw >> 16) & 0xFFFFu) | ((tdim1 & 0xFFFFu) << 16));
    g1[3] = (int)(((tdim1 >> 16) & 0xFFFFu) | ((tile0_dw & 0xFFFFu) << 16));
    g1[4] = (int)(tile1 & 0xFFFFu);                        // tile_dim1; tile_dim2=0 (2D)
    g1[5] = (int)stride_dw;                                // tensor_dim0_stride[31:0]
    g1[6] = 0;
    g1[7] = 0;
    i32x4 z4 = {0, 0, 0, 0};
    i32x8 z8 = {0, 0, 0, 0, 0, 0, 0, 0};
    __builtin_amdgcn_tensor_load_to_lds(g0, g1, z4, z4, z8, 0);
}

// ---------------- zero workspace ----------------
__global__ void zero_f32(float* __restrict__ p, size_t n) {
    size_t i = (size_t)blockIdx.x * blockDim.x + threadIdx.x;
    if (i < n) p[i] = 0.0f;
}

// ---------------- degree ----------------
__global__ void degree_kernel(const long long* __restrict__ ei, float* __restrict__ deg) {
    int e = blockIdx.x * blockDim.x + threadIdx.x;
    if (e < NE) {
        int dst = (int)ei[NE + e];
        atomicAdd(&deg[dst], 1.0f);
    }
}

// ---------------- invdeg (in place) ----------------
__global__ void invdeg_kernel(float* __restrict__ deg) {
    int n = blockIdx.x * blockDim.x + threadIdx.x;
    if (n < NN) {
        float d = deg[n];
        deg[n] = (d > 0.0f) ? (1.0f / d) : 0.0f;
    }
}

// ---------------- scatter-add: one wave per edge, lane owns a float4 ----------------
__global__ void scatter_kernel(const long long* __restrict__ ei,
                               const float* __restrict__ X, const float* __restrict__ H,
                               float* __restrict__ aggX, float* __restrict__ aggH) {
    size_t t = (size_t)blockIdx.x * blockDim.x + threadIdx.x;
    size_t e = t >> 5;
    int q = (int)(t & 31);
    if (e >= NE) return;
    int src = (int)ei[e];           // wave-uniform
    int dst = (int)ei[NE + e];      // wave-uniform
    const float4 xv = *(const float4*)(X + (size_t)src * 128 + q * 4);
    const float4 hv = *(const float4*)(H + (size_t)src * 128 + q * 4);
    float* ax = aggX + (size_t)dst * 128 + q * 4;
    float* ah = aggH + (size_t)dst * 128 + q * 4;
    atomicAdd(ax + 0, xv.x); atomicAdd(ax + 1, xv.y);
    atomicAdd(ax + 2, xv.z); atomicAdd(ax + 3, xv.w);
    atomicAdd(ah + 0, hv.x); atomicAdd(ah + 1, hv.y);
    atomicAdd(ah + 2, hv.z); atomicAdd(ah + 3, hv.w);
}

// ---------------- pre-convert activations to bf16: Ab = [agg*invdeg | root], [N,256] ----------------
__global__ void prep_act(const float* __restrict__ aggX, const float* __restrict__ aggH,
                         const float* __restrict__ X, const float* __restrict__ H,
                         const float* __restrict__ invdeg,
                         __bf16* __restrict__ Abx, __bf16* __restrict__ Abh) {
    size_t t = (size_t)blockIdx.x * blockDim.x + threadIdx.x;
    if (t >= (size_t)NN * 32) return;
    size_t n = t >> 5;
    int base = (int)(t & 31) * 8;
    float sc;
    const float *sx, *sh;
    if (base < 128) {
        sc = invdeg[n];
        sx = aggX + n * 128 + base;
        sh = aggH + n * 128 + base;
    } else {
        sc = 1.0f;
        sx = X + n * 128 + base - 128;
        sh = H + n * 128 + base - 128;
    }
    float4 x0 = *(const float4*)sx, x1 = *(const float4*)(sx + 4);
    float4 h0 = *(const float4*)sh, h1 = *(const float4*)(sh + 4);
    v8bf vx, vh;
    vx[0] = (__bf16)(x0.x * sc); vx[1] = (__bf16)(x0.y * sc);
    vx[2] = (__bf16)(x0.z * sc); vx[3] = (__bf16)(x0.w * sc);
    vx[4] = (__bf16)(x1.x * sc); vx[5] = (__bf16)(x1.y * sc);
    vx[6] = (__bf16)(x1.z * sc); vx[7] = (__bf16)(x1.w * sc);
    vh[0] = (__bf16)(h0.x * sc); vh[1] = (__bf16)(h0.y * sc);
    vh[2] = (__bf16)(h0.z * sc); vh[3] = (__bf16)(h0.w * sc);
    vh[4] = (__bf16)(h1.x * sc); vh[5] = (__bf16)(h1.y * sc);
    vh[6] = (__bf16)(h1.z * sc); vh[7] = (__bf16)(h1.w * sc);
    *(v8bf*)(Abx + n * 256 + base) = vx;
    *(v8bf*)(Abh + n * 256 + base) = vh;
}

// ---------------- pre-convert weights to bf16 B^T image: Wb[p][g][c][k], k in 0..255 ----------------
__global__ void prep_w(const float* __restrict__ Wl_x, const float* __restrict__ Wr_x,
                       const float* __restrict__ Wl_h, const float* __restrict__ Wr_h,
                       __bf16* __restrict__ Wb) {
    int t = blockIdx.x * blockDim.x + threadIdx.x;
    if (t >= 2 * 4 * 128 * 32) return;
    int k = (t & 31) * 8;
    int c = (t >> 5) & 127;
    int g = (t >> 12) & 3;
    int p = t >> 14;
    const float* Wl = p ? Wl_h : Wl_x;
    const float* Wr = p ? Wr_h : Wr_x;
    const float* src = (k < 128) ? (Wl + ((size_t)g * 128 + c) * 128 + k)
                                 : (Wr + ((size_t)g * 128 + c) * 128 + k - 128);
    float4 w0 = *(const float4*)src, w1 = *(const float4*)(src + 4);
    v8bf v;
    v[0] = (__bf16)w0.x; v[1] = (__bf16)w0.y; v[2] = (__bf16)w0.z; v[3] = (__bf16)w0.w;
    v[4] = (__bf16)w1.x; v[5] = (__bf16)w1.y; v[6] = (__bf16)w1.z; v[7] = (__bf16)w1.w;
    *(v8bf*)(Wb + ((((size_t)p * 4 + g) * 128 + c) * 256) + k) = v;
}

// ---------------- fused SAGE4 GEMM + bias + row-L2-normalize (TDM-staged bf16 WMMA) ------------
// Block: 256 threads (8 waves). Tile: 128 nodes x 128 cols for gate g = blockIdx.y.
// Wave 0 TDM-loads the A tile (128x32 bf16, hw-padded to LDA=40) and B^T tile per k-step.
__global__ __launch_bounds__(256, 1)
void sage4_gemm(const __bf16* __restrict__ Abx, const __bf16* __restrict__ Abh,
                const __bf16* __restrict__ Wb,
                const float* __restrict__ bl_x, const float* __restrict__ bl_h,
                float* __restrict__ S) {
    __shared__ __align__(16) __bf16 a_lds[128 * LDA];
    __shared__ __align__(16) __bf16 b_lds[128 * LDA];

    const int g     = blockIdx.y;
    const int node0 = blockIdx.x << 7;
    const int tid   = threadIdx.x;
    const int wave  = tid >> 5;
    const int lane  = tid & 31;
    const int l15   = lane & 15;
    const int hl    = lane >> 4;       // lane half: 0 or 1
    const int kbase = hl << 3;         // bf16 A/B fragment K base per ISA layout

    const unsigned lds_a = (unsigned)(uintptr_t)(void*)a_lds;  // LDS offset = addr[31:0]
    const unsigned lds_b = (unsigned)(uintptr_t)(void*)b_lds;
    const unsigned rows_left = (unsigned)(NN - node0);         // OOB rows -> zero fill

    v8f acc2[2][8];
#pragma unroll
    for (int p = 0; p < 2; ++p)
#pragma unroll
        for (int t = 0; t < 8; ++t)
#pragma unroll
            for (int i = 0; i < 8; ++i)
                acc2[p][t][i] = 0.0f;

#pragma unroll
    for (int pass = 0; pass < 2; ++pass) {
        const __bf16* Ab = pass ? Abh : Abx;
        const __bf16* Bb = Wb + (((size_t)pass * 4 + g) * 128) * 256;

        for (int ks = 0; ks < 8; ++ks) {             // K = 256 in steps of 32
            __syncthreads();                         // prior LDS readers done
            if (tid < 32) {
                const __bf16* Ag = Ab + (size_t)node0 * 256 + ks * 32;
                const __bf16* Bg = Bb + ks * 32;
                // 2D tiles of 16 DW x 128 rows; row stride 128 DW; LDS pad 4 DW per 16 DW
                tdm_load_2d(Ag, lds_a, 128u, rows_left, 16u, 128u, 128u);
                tdm_load_2d(Bg, lds_b, 128u, 128u, 16u, 128u, 128u);
                __builtin_amdgcn_s_wait_tensorcnt(0);
            }
            __syncthreads();                         // LDS tiles visible to all waves

            // A fragment (16x32 bf16): rows wave*16 + l15, K per ISA 16-bit A layout
            v16bf afrag;
            {
                const __bf16* pr = a_lds + (wave * 16 + l15) * LDA;
                v8bf lo = *(const v8bf*)(pr + kbase);
                v8bf hi = *(const v8bf*)(pr + kbase + 16);
                afrag = __builtin_shufflevector(lo, hi,
                        0, 1, 2, 3, 4, 5, 6, 7, 8, 9, 10, 11, 12, 13, 14, 15);
            }
#pragma unroll
            for (int t = 0; t < 8; ++t) {
                const __bf16* pc = b_lds + (t * 16 + l15) * LDA;
                v8bf lo = *(const v8bf*)(pc + kbase);
                v8bf hi = *(const v8bf*)(pc + kbase + 16);
                v16bf bfrag = __builtin_shufflevector(lo, hi,
                        0, 1, 2, 3, 4, 5, 6, 7, 8, 9, 10, 11, 12, 13, 14, 15);
                acc2[pass][t] = __builtin_amdgcn_wmma_f32_16x16x32_bf16(
                    false, afrag, false, bfrag, (short)0, acc2[pass][t], false, false);
            }
        }

        // bias add then per-row L2 normalize (row r lives in VGPR r%8 across a 16-lane half)
        const float* bl = (pass ? bl_h : bl_x) + g * 128;
        float ssq[8];
#pragma unroll
        for (int i = 0; i < 8; ++i) ssq[i] = 0.0f;
#pragma unroll
        for (int t = 0; t < 8; ++t) {
            const float bv = bl[t * 16 + l15];
#pragma unroll
            for (int i = 0; i < 8; ++i) {
                float v = acc2[pass][t][i] + bv;
                acc2[pass][t][i] = v;
                ssq[i] += v * v;
            }
        }
#pragma unroll
        for (int i = 0; i < 8; ++i) {
            float s = ssq[i];
            s += __shfl_xor(s, 1, 16);
            s += __shfl_xor(s, 2, 16);
            s += __shfl_xor(s, 4, 16);
            s += __shfl_xor(s, 8, 16);
            const float sc = 1.0f / fmaxf(sqrtf(s), 1e-12f);
#pragma unroll
            for (int t = 0; t < 8; ++t) acc2[pass][t][i] *= sc;
        }
    }

    // store s = normalize(x-path) + normalize(h-path)
    const size_t sbase = (size_t)g * NN * 128;
#pragma unroll
    for (int t = 0; t < 8; ++t) {
        const int col = t * 16 + l15;
#pragma unroll
        for (int i = 0; i < 8; ++i) {
            const int node = node0 + wave * 16 + (hl << 3) + i;
            if (node < NN)
                S[sbase + (size_t)node * 128 + col] = acc2[0][t][i] + acc2[1][t][i];
        }
    }
}

// ---------------- LSTM gate math, fully elementwise ----------------
__global__ void gate_finalize(const float* __restrict__ S, const float* __restrict__ C,
                              const float* __restrict__ w_c, const float* __restrict__ b_g,
                              float* __restrict__ out) {
    const size_t t = (size_t)blockIdx.x * blockDim.x + threadIdx.x;
    if (t >= (size_t)NN * 32) return;
    const size_t n  = t >> 5;
    const int    f  = (int)(t & 31) << 2;
    const size_t nf = n * 128 + f;
    const size_t NS = (size_t)NN * 128;

    float4 s0 = *(const float4*)(S + nf);
    float4 s1 = *(const float4*)(S + NS + nf);
    float4 s2 = *(const float4*)(S + 2 * NS + nf);
    float4 s3 = *(const float4*)(S + 3 * NS + nf);
    float4 c  = *(const float4*)(C + nf);
    float4 w0 = *(const float4*)(w_c + f);
    float4 w1 = *(const float4*)(w_c + 128 + f);
    float4 w2 = *(const float4*)(w_c + 256 + f);
    float4 b0 = *(const float4*)(b_g + f);
    float4 b1 = *(const float4*)(b_g + 128 + f);
    float4 b2 = *(const float4*)(b_g + 256 + f);
    float4 b3 = *(const float4*)(b_g + 384 + f);

    float4 h2, c2;
#define GATE(cmp) {                                              \
        float I  = sigmoidf(s0.cmp + w0.cmp * c.cmp + b0.cmp);   \
        float F  = sigmoidf(s1.cmp + w1.cmp * c.cmp + b1.cmp);   \
        float T  = tanhf(s2.cmp + b2.cmp);                       \
        float Cn = F * c.cmp + I * T;                            \
        float O  = sigmoidf(s3.cmp + w2.cmp * Cn + b3.cmp);      \
        c2.cmp = Cn; h2.cmp = O * tanhf(Cn); }
    GATE(x) GATE(y) GATE(z) GATE(w)
#undef GATE

    *(float4*)(out + nf)      = h2;
    *(float4*)(out + NS + nf) = c2;
}

extern "C" void kernel_launch(void* const* d_in, const int* in_sizes, int n_in,
                              void* d_out, int out_size, void* d_ws, size_t ws_size,
                              hipStream_t stream) {
    (void)in_sizes; (void)n_in; (void)out_size; (void)ws_size;

    const float*     X    = (const float*)d_in[0];
    const float*     H    = (const float*)d_in[1];
    const float*     C    = (const float*)d_in[2];
    const long long* ei   = (const long long*)d_in[3];
    const float*     Wl_x = (const float*)d_in[4];
    const float*     bl_x = (const float*)d_in[5];
    const float*     Wr_x = (const float*)d_in[6];
    const float*     Wl_h = (const float*)d_in[7];
    const float*     bl_h = (const float*)d_in[8];
    const float*     Wr_h = (const float*)d_in[9];
    const float*     w_c  = (const float*)d_in[10];
    const float*     b_g  = (const float*)d_in[11];
    float*           out  = (float*)d_out;

    // workspace layout (float units): deg | aggX | aggH | S[4,N,128] | Abx | Abh | Wb (bf16)
    float* ws = (float*)d_ws;
    size_t o = 0;
    float* deg  = ws;                 o = ((size_t)NN + 255) & ~(size_t)255;
    float* aggX = ws + o;             o += (size_t)NN * 128;
    float* aggH = ws + o;             size_t zend = o + (size_t)NN * 128;
    o = (zend + 255) & ~(size_t)255;
    float* S    = ws + o;             o += (size_t)4 * NN * 128;
    __bf16* Abx = (__bf16*)(ws + o);  o += (size_t)NN * 128;   // N*256 bf16 = N*128 floats
    __bf16* Abh = (__bf16*)(ws + o);  o += (size_t)NN * 128;
    __bf16* Wb  = (__bf16*)(ws + o);                            // 2*4*128*256 bf16

    // 1) zero deg + agg buffers
    zero_f32<<<(unsigned)((zend + 255) / 256), 256, 0, stream>>>(ws, zend);
    // 2) degrees
    degree_kernel<<<(NE + 255) / 256, 256, 0, stream>>>(ei, deg);
    // 3) scatter-add X and H features (wave-per-edge, atomics into L2-resident agg)
    scatter_kernel<<<(unsigned)(((size_t)NE * 32 + 255) / 256), 256, 0, stream>>>(
        ei, X, H, aggX, aggH);
    // 4) invdeg
    invdeg_kernel<<<(NN + 255) / 256, 256, 0, stream>>>(deg);
    // 5) bf16 pre-conversion of activations (invdeg fused) and weights
    prep_act<<<(unsigned)(((size_t)NN * 32 + 255) / 256), 256, 0, stream>>>(
        aggX, aggH, X, H, deg, Abx, Abh);
    prep_w<<<(2 * 4 * 128 * 32) / 256, 256, 0, stream>>>(Wl_x, Wr_x, Wl_h, Wr_h, Wb);
    // 6) fused SAGE4 GEMM (TDM-staged bf16 WMMA) + bias + row L2 normalize -> S[4,N,128]
    dim3 ggrid((NN + 127) / 128, 4);
    sage4_gemm<<<ggrid, 256, 0, stream>>>(Abx, Abh, Wb, bl_x, bl_h, S);
    // 7) gate math -> out = [H2 | C2]
    gate_finalize<<<(unsigned)(((size_t)NN * 32 + 255) / 256), 256, 0, stream>>>(
        S, C, w_c, b_g, out);
}
